// DinAttention_31396210933840
// MI455X (gfx1250) — matrix operand order repaired and checked by reading, not compiled, not run on hardware
//
#include <hip/hip_runtime.h>
#include <hip/hip_bf16.h>

typedef __attribute__((ext_vector_type(16))) _Float16 v16h;
typedef __attribute__((ext_vector_type(4)))  _Float16 v4h;
typedef __attribute__((ext_vector_type(2)))  _Float16 v2h;
typedef __attribute__((ext_vector_type(8)))  float    v8f;

#define L_SEQ   200
#define L_PAD   208      // 13 tiles of 16
#define MT_CNT  13
#define DDIM    64
#define F_IN    256      // 4*D
#define H1DIM   128
#define H2DIM   64

// ---- dynamic LDS layout (bytes) ----
#define OFF_W1F   0                         // 8*8 frags * 32 lanes * 8 dw * 4B = 65536
#define OFF_H1F   (OFF_W1F + 65536)         // per-wave: 8 waves * 4 frags * 32 * 8 dw *4 = 32768
#define OFF_W2F   (OFF_H1F + 32768)         // 4*4 frags * 32 * 8 dw * 4 = 16384
#define OFF_K     (OFF_W2F + 16384)         // 208*64 f16 = 26624
#define OFF_Q     (OFF_K + 26624)           // 64 f16 = 128
#define OFF_B1    (OFF_Q + 128)             // 128 f32 = 512
#define OFF_B2    (OFF_B1 + 512)            // 64 f32 = 256
#define OFF_W3    (OFF_B2 + 256)            // 64 f32 = 256
#define OFF_WV    (OFF_W3 + 256)            // 208 f32 = 832
#define OFF_OUTP  (OFF_WV + 832)            // 4*64 f32 = 1024
#define SMEM_BYTES (OFF_OUTP + 1024)        // = 144320

__device__ __forceinline__ float fast_sigmoid(float x) {
    return 1.0f / (1.0f + __expf(-x));
}

__global__ void __launch_bounds__(256, 1)
din_attention_kernel(const float* __restrict__ seq_g,   // (BS, 200, 64)
                     const float* __restrict__ query_g, // (BS, 64)
                     const int*   __restrict__ seqlen_g,// (BS,)
                     const float* __restrict__ W1,      // (256, 128)
                     const float* __restrict__ b1,      // (128,)
                     const float* __restrict__ W2,      // (128, 64)
                     const float* __restrict__ b2,      // (64,)
                     const float* __restrict__ W3,      // (64,)
                     const float* __restrict__ b3,      // (1,)
                     float*       __restrict__ out_g)   // (BS, 64)
{
    extern __shared__ char smem[];
    _Float16* sh_w1f = (_Float16*)(smem + OFF_W1F);
    _Float16* sh_h1f = (_Float16*)(smem + OFF_H1F);
    _Float16* sh_w2f = (_Float16*)(smem + OFF_W2F);
    _Float16* sh_k   = (_Float16*)(smem + OFF_K);
    _Float16* sh_q   = (_Float16*)(smem + OFF_Q);
    float*    sh_b1  = (float*)(smem + OFF_B1);
    float*    sh_b2  = (float*)(smem + OFF_B2);
    float*    sh_w3  = (float*)(smem + OFF_W3);
    float*    sh_w   = (float*)(smem + OFF_WV);
    float*    sh_outp= (float*)(smem + OFF_OUTP);

    const int tid  = threadIdx.x;
    const int lane = tid & 31;
    const int wid  = tid >> 5;
    const int laneM = lane & 15;
    const int laneH = lane >> 4;

    const int b = blockIdx.x;
    const float* seq = seq_g + (size_t)b * L_SEQ * DDIM;
    const float* qv  = query_g + (size_t)b * DDIM;
    const int slen   = min(seqlen_g[b], L_SEQ);
    const float b3v  = b3[0];

    // ---------------- Phase 0: cooperative LDS fill ----------------
    // query -> f16
    if (tid < DDIM) sh_q[tid] = (_Float16)qv[tid];
    // biases / W3
    if (tid < H1DIM) sh_b1[tid] = b1[tid];
    if (tid < H2DIM) { sh_b2[tid] = b2[tid]; sh_w3[tid] = W3[tid]; }

    // sequence tile -> f16 (pad rows 200..207 with zeros), float4 granularity
    for (int i = tid; i < L_PAD * (DDIM / 4); i += 256) {
        int l  = i >> 4;
        int d4 = (i & 15) << 2;
        float4 v;
        if (l < L_SEQ) v = *(const float4*)(seq + l * DDIM + d4);
        else           v = make_float4(0.f, 0.f, 0.f, 0.f);
        v4h p = { (_Float16)v.x, (_Float16)v.y, (_Float16)v.z, (_Float16)v.w };
        *(v4h*)(sh_k + l * DDIM + d4) = p;
    }

    // W1 -> f16, pre-swizzled to WMMA B-fragment layout.
    // B(32x16) layout: lane = n + 16*((k/8)&1); VGPR v = (k/16)*4 + (k%8)/2; half = k&1.
    for (int i = tid; i < F_IN * H1DIM; i += 256) {
        int k = i >> 7, n = i & 127;
        _Float16 val = (_Float16)W1[i];
        int kt = k >> 5, nt = n >> 4, kloc = k & 31, nloc = n & 15;
        int ln = nloc + (((kloc >> 3) & 1) << 4);
        int vv = ((kloc >> 4) << 2) + ((kloc & 7) >> 1), hi = kloc & 1;
        sh_w1f[(((kt * 8 + nt) * 32 + ln) * 8 + vv) * 2 + hi] = val;
    }
    // W2 -> f16 fragments (4 kt x 4 nt)
    for (int i = tid; i < H1DIM * H2DIM; i += 256) {
        int k = i >> 6, n = i & 63;
        _Float16 val = (_Float16)W2[i];
        int kt = k >> 5, nt = n >> 4, kloc = k & 31, nloc = n & 15;
        int ln = nloc + (((kloc >> 3) & 1) << 4);
        int vv = ((kloc >> 4) << 2) + ((kloc & 7) >> 1), hi = kloc & 1;
        sh_w2f[(((kt * 4 + nt) * 32 + ln) * 8 + vv) * 2 + hi] = val;
    }
    __syncthreads();

    // ---------------- Phase 1: per-M-tile pipeline ----------------
    for (int mt = wid; mt < MT_CNT; mt += 8) {
        // ---- GEMM1: (16 x 256) @ (256 x 128) ----
        v8f acc[8];
        #pragma unroll
        for (int nt = 0; nt < 8; ++nt) acc[nt] = (v8f)(0.0f);

        const int row = mt * 16 + laneM;
        #pragma unroll
        for (int kt = 0; kt < 8; ++kt) {
            const int sector = kt >> 1;          // 0:q 1:k 2:q-k 3:q*k
            const int dbase  = (kt & 1) * 32;
            // Build A fragment (16x32 f16) on the fly.
            // A layout: lane = m + 16*half; VGPR v: K = (v/4)*16 + half*8 + (v%4)*2
            v16h a;
            #pragma unroll
            for (int v = 0; v < 8; ++v) {
                int kloc = ((v >> 2) << 4) + (laneH << 3) + ((v & 3) << 1);
                int d = dbase + kloc;
                v2h kp = *(const v2h*)(sh_k + row * DDIM + d);
                v2h qp = *(const v2h*)(sh_q + d);
                v2h val;
                if      (sector == 0) val = qp;
                else if (sector == 1) val = kp;
                else if (sector == 2) val = qp - kp;
                else                  val = qp * kp;
                a[2 * v]     = val.x;
                a[2 * v + 1] = val.y;
            }
            #pragma unroll
            for (int nt = 0; nt < 8; ++nt) {
                v16h bf = *(const v16h*)(sh_w1f + ((kt * 8 + nt) * 32 + lane) * 16);
                acc[nt] = __builtin_amdgcn_wmma_f32_16x16x32_f16(
                    false, a, false, bf, (short)0, acc[nt], false, false);
            }
        }

        // ---- bias + sigmoid, write h1 in A-fragment layout (per-wave slot) ----
        #pragma unroll
        for (int nt = 0; nt < 8; ++nt) {
            const int g   = nt * 16 + laneM;
            const float bg = sh_b1[g];
            const int kt2 = nt >> 1;
            const int hh  = laneM >> 3;
            const int kk  = laneM & 7;
            const int vv  = (nt & 1) * 4 + (kk >> 1);
            const int hi  = kk & 1;
            #pragma unroll
            for (int r = 0; r < 8; ++r) {
                int m = r + (laneH << 3);          // D layout: lanes16-31 hold M=r+8
                float s = fast_sigmoid(acc[nt][r] + bg);
                int lane_t = m + 16 * hh;
                sh_h1f[(((wid * 4 + kt2) * 32 + lane_t) * 8 + vv) * 2 + hi] = (_Float16)s;
            }
        }
        // same-wave LDS ops are in-order: no barrier needed before reading back.

        // ---- GEMM2: (16 x 128) @ (128 x 64) ----
        v8f acc2[4];
        #pragma unroll
        for (int nt = 0; nt < 4; ++nt) acc2[nt] = (v8f)(0.0f);
        #pragma unroll
        for (int kt = 0; kt < 4; ++kt) {
            v16h a2 = *(const v16h*)(sh_h1f + ((wid * 4 + kt) * 32 + lane) * 16);
            #pragma unroll
            for (int nt = 0; nt < 4; ++nt) {
                v16h bf = *(const v16h*)(sh_w2f + ((kt * 4 + nt) * 32 + lane) * 16);
                acc2[nt] = __builtin_amdgcn_wmma_f32_16x16x32_f16(
                    false, a2, false, bf, (short)0, acc2[nt], false, false);
            }
        }

        // ---- bias + sigmoid + dot with W3 (VALU), reduce over 16 lanes ----
        float wpart[8];
        #pragma unroll
        for (int r = 0; r < 8; ++r) wpart[r] = 0.0f;
        #pragma unroll
        for (int nt = 0; nt < 4; ++nt) {
            int g = nt * 16 + laneM;
            float b2g = sh_b2[g];
            float w3g = sh_w3[g];
            #pragma unroll
            for (int r = 0; r < 8; ++r) {
                float s = fast_sigmoid(acc2[nt][r] + b2g);
                wpart[r] += s * w3g;
            }
        }
        #pragma unroll
        for (int r = 0; r < 8; ++r) {
            float t = wpart[r];
            t += __shfl_xor(t, 1, 16);
            t += __shfl_xor(t, 2, 16);
            t += __shfl_xor(t, 4, 16);
            t += __shfl_xor(t, 8, 16);
            if (laneM == 0) {
                int l = mt * 16 + r + (laneH << 3);
                sh_w[l] = (l < slen) ? (t + b3v) : 0.0f;
            }
        }
    }
    __syncthreads();

    // ---------------- Phase 2: out[b,d] = sum_l w[l] * k[l,d] (f32 from global) ----
    {
        int d = tid & 63;
        int grp = tid >> 6;       // 4 groups of 64 threads
        float accd = 0.0f;
        for (int l = grp; l < slen; l += 4)
            accd += sh_w[l] * seq[l * DDIM + d];
        sh_outp[grp * 64 + d] = accd;
    }
    __syncthreads();
    if (tid < 64) {
        float r = sh_outp[tid] + sh_outp[64 + tid] + sh_outp[128 + tid] + sh_outp[192 + tid];
        out_g[(size_t)b * DDIM + tid] = r;
    }
}

extern "C" void kernel_launch(void* const* d_in, const int* in_sizes, int n_in,
                              void* d_out, int out_size, void* d_ws, size_t ws_size,
                              hipStream_t stream) {
    const float* seq    = (const float*)d_in[0];
    const float* query  = (const float*)d_in[1];
    const int*   seqlen = (const int*)d_in[2];
    const float* W1     = (const float*)d_in[3];
    const float* b1     = (const float*)d_in[4];
    const float* W2     = (const float*)d_in[5];
    const float* b2     = (const float*)d_in[6];
    const float* W3     = (const float*)d_in[7];
    const float* b3     = (const float*)d_in[8];
    float* out          = (float*)d_out;

    const int BS = in_sizes[2];   // one seqlen entry per batch row

    din_attention_kernel<<<dim3(BS), dim3(256), SMEM_BYTES, stream>>>(
        seq, query, seqlen, W1, b1, W2, b2, W3, b3, out);
}
